// LatentMoELayer_2877628088578
// MI455X (gfx1250) — matrix operand order repaired and checked by reading, not compiled, not run on hardware
//
#include <hip/hip_runtime.h>
#include <hip/hip_bf16.h>

// ---------------- types ----------------
typedef __attribute__((ext_vector_type(16))) _Float16 v16h;
typedef __attribute__((ext_vector_type(8)))  _Float16 v8h;
typedef __attribute__((ext_vector_type(8)))  float    v8f;

// ---------------- problem constants ----------------
constexpr int Tt  = 4096;   // B*S tokens
constexpr int Dd  = 1024;   // d_model
constexpr int Ll  = 256;    // d_latent
constexpr int Hh  = 512;    // expert hidden
constexpr int Ee  = 8;      // experts
constexpr int SHs = 512;    // shared hidden

// ---------------- tiling ----------------
constexpr int MBLK  = 128;              // block tile M
constexpr int NBLK  = 128;              // block tile N
constexpr int KSTEP = 32;               // K per WMMA step
constexpr int ROWB  = 80;               // LDS row stride bytes (64B data + 16B pad)
constexpr int TILEB = MBLK * ROWB;      // 10240 B per staged tile

// ---------------- WMMA helper ----------------
__device__ __forceinline__ v8f wmma16(v16h a, v16h b, v8f c) {
  return __builtin_amdgcn_wmma_f32_16x16x32_f16(false, a, false, b, (short)0, c, false, false);
}

// ---------------- async global -> LDS (gfx1250, ASYNCcnt) ----------------
__device__ __forceinline__ unsigned lds_off(const void* p) {
  // generic LDS address: addr[31:0] is the LDS byte offset
  return (unsigned)(size_t)p;
}
__device__ __forceinline__ void async_cp16(unsigned loff, const void* g) {
  asm volatile("global_load_async_to_lds_b128 %0, %1, off"
               :: "v"(loff), "v"(g) : "memory");
}
template <int N> __device__ __forceinline__ void wait_async() {
  asm volatile("s_wait_asynccnt %0" :: "n"(N) : "memory");
}

// stage one MBLKx32-half tile (rows of 64B) into LDS, 16B chunks, 2 async ops per thread
__device__ __forceinline__ void stage_tile(const _Float16* __restrict__ G, int ldg,
                                           int row0, int k0, char* tile, int tid) {
  {
    int row = tid >> 2, col = tid & 3;
    async_cp16(lds_off(tile + row * ROWB + col * 16),
               G + (size_t)(row0 + row) * ldg + k0 + col * 8);
  }
  {
    int c = tid + 256;
    int row = c >> 2, col = c & 3;
    async_cp16(lds_off(tile + row * ROWB + col * 16),
               G + (size_t)(row0 + row) * ldg + k0 + col * 8);
  }
}

// ---------------- fragment loads from LDS ----------------
// A frag 16x32: lanes 0-15: M=lane, K=0..15; lanes 16-31: M=lane-16, K=16..31
__device__ __forceinline__ v16h lds_a_frag(const char* tile, int m0, int lane) {
  int m  = m0 + (lane & 15);
  int kb = (lane < 16) ? 0 : 16;
  const _Float16* p = (const _Float16*)(tile + m * ROWB) + kb;
  v8h lo = *(const v8h*)(p);
  v8h hi = *(const v8h*)(p + 8);
  return __builtin_shufflevector(lo, hi, 0,1,2,3,4,5,6,7,8,9,10,11,12,13,14,15);
}
// B frag 32x16 from (N,K)-major rows: lanes 0-15: N=lane, K=0..15; lanes 16-31: N=lane-16, K=16..31
__device__ __forceinline__ v16h lds_b_frag(const char* tile, int n0, int lane) {
  int n  = n0 + (lane & 15);
  int kk = (lane < 16) ? 0 : 16;
  return *(const v16h*)((const _Float16*)(tile + n * ROWB) + kk);
}

// ---------------- fp32 -> fp16 convert ----------------
__global__ void cvt_f32_f16(const float* __restrict__ in, _Float16* __restrict__ out, int n) {
  int i = blockIdx.x * blockDim.x + threadIdx.x;
  if (i < n) out[i] = (_Float16)in[i];
}

// ---------------- router: logits -> top2 softmax -> dense comb (T,E) ----------------
__global__ __launch_bounds__(256) void router_kernel(const float* __restrict__ x,
                                                     const float* __restrict__ w_router,
                                                     float* __restrict__ comb) {
  int lane = threadIdx.x & 31;
  int wid  = threadIdx.x >> 5;
  int t = blockIdx.x * 8 + wid;
  if (t >= Tt) return;
  const float* xr = x + (size_t)t * Dd;
  float acc[Ee];
  #pragma unroll
  for (int e = 0; e < Ee; ++e) acc[e] = 0.f;
  for (int d = lane; d < Dd; d += 32) {
    float xv = xr[d];
    #pragma unroll
    for (int e = 0; e < Ee; ++e) acc[e] += xv * w_router[(size_t)e * Dd + d];
  }
  #pragma unroll
  for (int e = 0; e < Ee; ++e)
    #pragma unroll
    for (int off = 16; off > 0; off >>= 1)
      acc[e] += __shfl_xor(acc[e], off, 32);
  if (lane == 0) {
    int i1 = 0; float v1 = acc[0];
    #pragma unroll
    for (int e = 1; e < Ee; ++e) if (acc[e] > v1) { v1 = acc[e]; i1 = e; }
    int i2 = -1; float v2 = -3.4e38f;
    #pragma unroll
    for (int e = 0; e < Ee; ++e) if (e != i1 && acc[e] > v2) { v2 = acc[e]; i2 = e; }
    float e2 = __expf(v2 - v1);          // v1 >= v2
    float inv = 1.f / (1.f + e2);
    float* row = comb + (size_t)t * Ee;
    #pragma unroll
    for (int e = 0; e < Ee; ++e) row[e] = 0.f;
    row[i1] = inv;
    row[i2] = e2 * inv;
  }
}

// ---------------- WMMA GEMM with double-buffered async LDS staging ----------------
// C(MxN) = A(MxK f16) * W(N,K f16)^T. Block 256thr = 8 waves (2M x 4N), wave tile 64x32.
// Main loop peeled: steady state always prefetches (branch-free), last step waits to 0.
__global__ __launch_bounds__(256) void gemm_wmma(const _Float16* __restrict__ A,
                                                 const _Float16* __restrict__ W,
                                                 float* __restrict__ Cf,
                                                 _Float16* __restrict__ Ch,
                                                 int M, int N, int K, int accum) {
  __shared__ __attribute__((aligned(32))) char smem[4 * TILEB]; // [A0,A1,B0,B1]

  int tid  = threadIdx.x;
  int lane = tid & 31;
  int wid  = tid >> 5;
  int mblk = blockIdx.y * MBLK;
  int nblk = blockIdx.x * NBLK;
  int wm = (wid & 1) * 64;    // wave M offset in block tile
  int wn = (wid >> 1) * 32;   // wave N offset in block tile

  v8f acc[4][2] = {};
  auto compute = [&](const char* Acur, const char* Bcur) {
    v16h af[4], bf[2];
    #pragma unroll
    for (int i = 0; i < 4; ++i) af[i] = lds_a_frag(Acur, wm + i * 16, lane);
    #pragma unroll
    for (int j = 0; j < 2; ++j) bf[j] = lds_b_frag(Bcur, wn + j * 16, lane);
    #pragma unroll
    for (int i = 0; i < 4; ++i)
      #pragma unroll
      for (int j = 0; j < 2; ++j)
        acc[i][j] = wmma16(af[i], bf[j], acc[i][j]);
  };

  stage_tile(A, K, mblk, 0, smem + 0 * TILEB, tid);
  stage_tile(W, K, nblk, 0, smem + 2 * TILEB, tid);

  int nks = K / KSTEP;                 // >= 2 for all our shapes
  for (int ks = 0; ks < nks - 1; ++ks) {
    int buf = ks & 1;
    stage_tile(A, K, mblk, (ks + 1) * KSTEP, smem + (0 + (buf ^ 1)) * TILEB, tid);
    stage_tile(W, K, nblk, (ks + 1) * KSTEP, smem + (2 + (buf ^ 1)) * TILEB, tid);
    wait_async<4>();                   // current buffer complete, prefetch in flight
    __syncthreads();
    compute(smem + (0 + buf) * TILEB, smem + (2 + buf) * TILEB);
    __syncthreads();
  }
  {
    int buf = (nks - 1) & 1;
    wait_async<0>();
    __syncthreads();
    compute(smem + (0 + buf) * TILEB, smem + (2 + buf) * TILEB);
  }

  int nlo = lane & 15;
  int mhi = (lane < 16) ? 0 : 8;
  #pragma unroll
  for (int i = 0; i < 4; ++i)
    #pragma unroll
    for (int j = 0; j < 2; ++j)
      #pragma unroll
      for (int r = 0; r < 8; ++r) {
        int m = mblk + wm + i * 16 + mhi + r;
        int n = nblk + wn + j * 16 + nlo;
        size_t idx = (size_t)m * N + n;
        float v = acc[i][j][r];
        if (accum) v += Cf[idx];
        if (Cf) Cf[idx] = v;
        if (Ch) Ch[idx] = (_Float16)v;
      }
}

// ---------------- fused dual-GEMM SwiGLU with async LDS staging ----------------
// Out(MxN f16) = silu(A Wg^T) * (A W1^T) * (comb[t,expert] or 1)
__global__ __launch_bounds__(256) void swiglu_wmma(const _Float16* __restrict__ A,
                                                   const _Float16* __restrict__ Wg,
                                                   const _Float16* __restrict__ W1,
                                                   _Float16* __restrict__ Out,
                                                   const float* __restrict__ comb,
                                                   int expert, int M, int N, int K) {
  __shared__ __attribute__((aligned(32))) char smem[6 * TILEB]; // [A0,A1,G0,G1,U0,U1]

  int tid  = threadIdx.x;
  int lane = tid & 31;
  int wid  = tid >> 5;
  int mblk = blockIdx.y * MBLK;
  int nblk = blockIdx.x * NBLK;
  int wm = (wid & 1) * 64;
  int wn = (wid >> 1) * 32;

  v8f gacc[4][2] = {};
  v8f aacc[4][2] = {};
  auto compute = [&](const char* Acur, const char* Gcur, const char* Ucur) {
    v16h af[4], bg[2], bu[2];
    #pragma unroll
    for (int i = 0; i < 4; ++i) af[i] = lds_a_frag(Acur, wm + i * 16, lane);
    #pragma unroll
    for (int j = 0; j < 2; ++j) {
      bg[j] = lds_b_frag(Gcur, wn + j * 16, lane);
      bu[j] = lds_b_frag(Ucur, wn + j * 16, lane);
    }
    #pragma unroll
    for (int i = 0; i < 4; ++i)
      #pragma unroll
      for (int j = 0; j < 2; ++j) {
        gacc[i][j] = wmma16(af[i], bg[j], gacc[i][j]);
        aacc[i][j] = wmma16(af[i], bu[j], aacc[i][j]);
      }
  };

  stage_tile(A,  K, mblk, 0, smem + 0 * TILEB, tid);
  stage_tile(Wg, K, nblk, 0, smem + 2 * TILEB, tid);
  stage_tile(W1, K, nblk, 0, smem + 4 * TILEB, tid);

  int nks = K / KSTEP;
  for (int ks = 0; ks < nks - 1; ++ks) {
    int buf = ks & 1;
    stage_tile(A,  K, mblk, (ks + 1) * KSTEP, smem + (0 + (buf ^ 1)) * TILEB, tid);
    stage_tile(Wg, K, nblk, (ks + 1) * KSTEP, smem + (2 + (buf ^ 1)) * TILEB, tid);
    stage_tile(W1, K, nblk, (ks + 1) * KSTEP, smem + (4 + (buf ^ 1)) * TILEB, tid);
    wait_async<6>();
    __syncthreads();
    compute(smem + (0 + buf) * TILEB, smem + (2 + buf) * TILEB, smem + (4 + buf) * TILEB);
    __syncthreads();
  }
  {
    int buf = (nks - 1) & 1;
    wait_async<0>();
    __syncthreads();
    compute(smem + (0 + buf) * TILEB, smem + (2 + buf) * TILEB, smem + (4 + buf) * TILEB);
  }

  int nlo = lane & 15;
  int mhi = (lane < 16) ? 0 : 8;
  #pragma unroll
  for (int i = 0; i < 4; ++i)
    #pragma unroll
    for (int r = 0; r < 8; ++r) {
      int m = mblk + wm + i * 16 + mhi + r;
      float scale = comb ? comb[(size_t)m * Ee + expert] : 1.0f;
      #pragma unroll
      for (int j = 0; j < 2; ++j) {
        int n = nblk + wn + j * 16 + nlo;
        float g = gacc[i][j][r];
        float a = aacc[i][j][r];
        float s = g / (1.f + __expf(-g));   // silu
        Out[(size_t)m * N + n] = (_Float16)(s * a * scale);
      }
    }
}

// ---------------- host side ----------------
static inline void cvt(const float* in, _Float16* out, int n, hipStream_t s) {
  cvt_f32_f16<<<(n + 255) / 256, 256, 0, s>>>(in, out, n);
}

extern "C" void kernel_launch(void* const* d_in, const int* in_sizes, int n_in,
                              void* d_out, int out_size, void* d_ws, size_t ws_size,
                              hipStream_t stream) {
  const float* x        = (const float*)d_in[0];   // (T, D)
  const float* w_router = (const float*)d_in[1];   // (E, D)
  const float* w_down   = (const float*)d_in[2];   // (L, D)
  const float* w_up     = (const float*)d_in[3];   // (D, L)
  const float* w1_e     = (const float*)d_in[4];   // (E, H, L)
  const float* wg_e     = (const float*)d_in[5];   // (E, H, L)
  const float* w2_e     = (const float*)d_in[6];   // (E, L, H)
  const float* w1_s     = (const float*)d_in[7];   // (SH, D)
  const float* wg_s     = (const float*)d_in[8];   // (SH, D)
  const float* w2_s     = (const float*)d_in[9];   // (D, SH)
  float* out = (float*)d_out;                      // (T, D)

  char* p = (char*)d_ws;
  auto take = [&](size_t bytes) { char* r = p; p += (bytes + 255) & ~(size_t)255; return r; };
  _Float16* x_h     = (_Float16*)take((size_t)Tt * Dd * 2);
  _Float16* wdown_h = (_Float16*)take((size_t)Ll * Dd * 2);
  _Float16* wup_h   = (_Float16*)take((size_t)Dd * Ll * 2);
  _Float16* w1e_h   = (_Float16*)take((size_t)Ee * Hh * Ll * 2);
  _Float16* wge_h   = (_Float16*)take((size_t)Ee * Hh * Ll * 2);
  _Float16* w2e_h   = (_Float16*)take((size_t)Ee * Ll * Hh * 2);
  _Float16* w1s_h   = (_Float16*)take((size_t)SHs * Dd * 2);
  _Float16* wgs_h   = (_Float16*)take((size_t)SHs * Dd * 2);
  _Float16* w2s_h   = (_Float16*)take((size_t)Dd * SHs * 2);
  float*    comb    = (float*)   take((size_t)Tt * Ee * 4);
  _Float16* lat_h   = (_Float16*)take((size_t)Tt * Ll * 2);
  _Float16* he_h    = (_Float16*)take((size_t)Tt * Hh * 2);
  float*    rl_f    = (float*)   take((size_t)Tt * Ll * 4);
  _Float16* rl_h    = (_Float16*)take((size_t)Tt * Ll * 2);
  _Float16* hs_h    = (_Float16*)take((size_t)Tt * SHs * 2);
  (void)ws_size; (void)n_in; (void)in_sizes; (void)out_size;

  // 1) precision conversion
  cvt(x,      x_h,     Tt * Dd,      stream);
  cvt(w_down, wdown_h, Ll * Dd,      stream);
  cvt(w_up,   wup_h,   Dd * Ll,      stream);
  cvt(w1_e,   w1e_h,   Ee * Hh * Ll, stream);
  cvt(wg_e,   wge_h,   Ee * Hh * Ll, stream);
  cvt(w2_e,   w2e_h,   Ee * Ll * Hh, stream);
  cvt(w1_s,   w1s_h,   SHs * Dd,     stream);
  cvt(wg_s,   wgs_h,   SHs * Dd,     stream);
  cvt(w2_s,   w2s_h,   Dd * SHs,     stream);

  // 2) router -> comb
  router_kernel<<<Tt / 8, 256, 0, stream>>>(x, w_router, comb);

  // 3) lat = x @ w_down^T  (T x L), f16 output
  gemm_wmma<<<dim3(Ll / NBLK, Tt / MBLK), 256, 0, stream>>>(
      x_h, wdown_h, nullptr, lat_h, Tt, Ll, Dd, 0);

  // 4) per-expert SwiGLU (comb folded in) + accumulate routed_lat (fp32)
  for (int e = 0; e < Ee; ++e) {
    swiglu_wmma<<<dim3(Hh / NBLK, Tt / MBLK), 256, 0, stream>>>(
        lat_h, wge_h + (size_t)e * Hh * Ll, w1e_h + (size_t)e * Hh * Ll,
        he_h, comb, e, Tt, Hh, Ll);
    gemm_wmma<<<dim3(Ll / NBLK, Tt / MBLK), 256, 0, stream>>>(
        he_h, w2e_h + (size_t)e * Ll * Hh, rl_f, nullptr, Tt, Ll, Hh, e > 0 ? 1 : 0);
  }
  cvt(rl_f, rl_h, Tt * Ll, stream);

  // 5) routed = routed_lat @ w_up^T  -> d_out (fp32, overwrite)
  gemm_wmma<<<dim3(Dd / NBLK, Tt / MBLK), 256, 0, stream>>>(
      rl_h, wup_h, out, nullptr, Tt, Dd, Ll, 0);

  // 6) shared SwiGLU in d_model, then d_out += hs @ w2_s^T
  swiglu_wmma<<<dim3(SHs / NBLK, Tt / MBLK), 256, 0, stream>>>(
      x_h, wgs_h, w1s_h, hs_h, nullptr, 0, Tt, SHs, Dd);
  gemm_wmma<<<dim3(Dd / NBLK, Tt / MBLK), 256, 0, stream>>>(
      hs_h, w2s_h, out, nullptr, Tt, Dd, SHs, 1);
}
